// arHOCA_50680614093334
// MI455X (gfx1250) — compile-verified, matrix operand order checked by reading, not run on hardware
//
#include <hip/hip_runtime.h>
#include <hip/hip_bf16.h>
#include <cstdint>

#define L  200
#define Q  21
#define H  128
#define B  256
#define LQ 4200

typedef __attribute__((ext_vector_type(16))) _Float16 v16h;
typedef __attribute__((ext_vector_type(8)))  float    v8f;
typedef __attribute__((ext_vector_type(4)))  unsigned int u32x4;
typedef __attribute__((ext_vector_type(8)))  int      i32x8;
typedef __attribute__((ext_vector_type(4)))  int      i32x4;

__device__ __forceinline__ float lrelu(float v) { return v >= 0.f ? v : 0.1f * v; }

__device__ __forceinline__ v8f v8f_zero() {
    v8f z;
#pragma unroll
    for (int i = 0; i < 8; ++i) z[i] = 0.f;
    return z;
}

// Build a 16x32 f16 WMMA fragment (A layout; B uses same pattern when the
// matrix is staged N-major / K-contiguous) from a row-major [rows][128] LDS
// tile. Per ISA 7.12.2: lane<16 -> M=lane, K pairs {0..7,16..23};
// lane>=16 -> M=lane-16, K pairs {8..15,24..31}. Two ds_read_b128 per frag.
__device__ __forceinline__ v16h frag_ld(const _Float16* base, int row0, int k0, int lane) {
    const int lr = lane & 15, hb = lane >> 4;
    const _Float16* r = base + (row0 + lr) * H + k0 + hb * 8;
    union { v16h v; uint4 q[2]; } u;
    u.q[0] = *(const uint4*)(r);
    u.q[1] = *(const uint4*)(r + 16);
    return u.v;
}

// ---------------- kernel 0: decode one-hot x -> idx[b,l] ----------------
__global__ void k_idx(const float* __restrict__ x, int* __restrict__ idx) {
    int t = blockIdx.x * blockDim.x + threadIdx.x;
    if (t >= B * L) return;
    const float* xp = x + (size_t)t * Q;
    int best = 0; float bv = xp[0];
#pragma unroll
    for (int q = 1; q < Q; ++q) { float v = xp[q]; if (v > bv) { bv = v; best = q; } }
    idx[t] = best;
}

// ---------------- kernel 1: h1 gather (one-hot x W1) ----------------
// h1[b,li,:] = lrelu(b1[li] + sum_{l<=li} W1[li][l*Q+idx[b,l], :]) stored f16.
__global__ void k_h1(const float* __restrict__ W1, const float* __restrict__ b1,
                     const int* __restrict__ idx, _Float16* __restrict__ h1o) {
    const int li = blockIdx.y;     // 0..L-2
    const int b  = blockIdx.x;     // 0..B-1
    const int h  = threadIdx.x;    // 0..H-1
    const float* Wl = W1 + (size_t)li * LQ * H;
    const int* ib = idx + b * L;
    float acc = b1[li * H + h];
    for (int l = 0; l <= li; ++l) {
        int c = ib[l];
        if (l + 1 <= li)
            __builtin_prefetch(Wl + (size_t)((l + 1) * Q + ib[l + 1]) * H + h, 0, 1);
        acc += Wl[(size_t)(l * Q + c) * H + h];
    }
    h1o[((size_t)li * B + b) * H + h] = (_Float16)lrelu(acc);
}

// -------- kernel 2: per-l fused WMMA FFN (h2 GEMM + out GEMM) + DCA epilogue --------
__global__ void __launch_bounds__(256, 1)
k_ffn(const _Float16* __restrict__ h1all,
      const float* __restrict__ W2, const float* __restrict__ b2,
      const float* __restrict__ Wout, const float* __restrict__ bout,
      const float* __restrict__ Wdca, const float* __restrict__ dca_bias,
      const int* __restrict__ idx, float* __restrict__ out) {
    extern __shared__ char smem[];
    _Float16* sh_a   = (_Float16*)(smem);                          // 256x128 f16 (h1, then h2)
    _Float16* sh_w2t = (_Float16*)(smem + 65536);                  // 128x128 f16, [n][k]
    _Float16* sh_wo  = (_Float16*)(smem + 65536 + 32768);          // 32x128 f16, [q][k], q>=21 zero
    float*    sh_b2  = (float*)(smem + 65536 + 32768 + 8192);      // 128
    float*    sh_bo  = (float*)(smem + 65536 + 32768 + 8192 + 512);// 32

    const int li   = blockIdx.x;   // 0..L-2
    const int p    = li + 1;       // output position
    const int tid  = threadIdx.x;
    const int lane = tid & 31;
    const int wid  = tid >> 5;
    const int lr   = lane & 15, hb = lane >> 4;

    // ---- stage W2[li] transposed f32->f16 into LDS ----
    const float* W2l = W2 + (size_t)li * H * H;
    for (int e = tid; e < H * H; e += 256) {
        int k = e >> 7, n = e & 127;                // W2l[k][n]
        sh_w2t[n * H + k] = (_Float16)W2l[e];
    }
    // ---- stage Wout[li] transposed, padded Q->32 ----
    const float* Wol = Wout + (size_t)li * H * Q;
    for (int e = tid; e < 32 * H; e += 256) {
        int q = e >> 7, k = e & 127;
        sh_wo[e] = (q < Q) ? (_Float16)Wol[(size_t)k * Q + q] : (_Float16)0.f;
    }
    if (tid < H) sh_b2[tid] = b2[li * H + tid];
    if (tid < 32) sh_bo[tid] = (tid < Q) ? bout[li * Q + tid] : 0.f;

    // ---- stage h1 tile (64 KB) global -> LDS via Tensor Data Mover ----
    const _Float16* h1l = h1all + (size_t)li * B * H;
#if defined(__gfx1250__) && __has_builtin(__builtin_amdgcn_tensor_load_to_lds)
    if (wid == 0) {
        const uint32_t lds_addr = (uint32_t)(uintptr_t)(void*)sh_a;  // low 32 bits = LDS offset
        const uint64_t ga = (uint64_t)(uintptr_t)(const void*)h1l;
        const uint32_t nelem = (B * H * 2) / 8;   // 8192 elements of 8 bytes
        u32x4 g0;
        g0.x = 1u;                                             // count=1 (valid descriptor)
        g0.y = lds_addr;                                       // lds_addr
        g0.z = (uint32_t)ga;                                   // global_addr[31:0]
        g0.w = ((uint32_t)(ga >> 32) & 0x01FFFFFFu) | 0x80000000u; // addr[56:32] | type=2
        i32x8 g1;
        g1[0] = (int)(3u << 16);                  // workgroup_mask=0, data_size=8B
        g1[1] = (int)((nelem & 0xFFFFu) << 16);   // tensor_dim0 low16 at bits[63:48]
        g1[2] = (int)((nelem >> 16) | (1u << 16));// tensor_dim0 hi16 | tensor_dim1=1 low16
        g1[3] = (int)((nelem & 0xFFFFu) << 16);   // tile_dim0 at bits[127:112]
        g1[4] = 0;                                // tile_dim1/2 unused
        g1[5] = (int)nelem;                       // tensor_dim0_stride low32
        g1[6] = 0;
        g1[7] = 0;
        i32x4 g2; i32x4 g3;
#pragma unroll
        for (int i = 0; i < 4; ++i) { g2[i] = 0; g3[i] = 0; }
#if __clang_major__ >= 23
        i32x8 z8;
#pragma unroll
        for (int i = 0; i < 8; ++i) z8[i] = 0;
        __builtin_amdgcn_tensor_load_to_lds(g0, g1, g2, g3, z8, 0);
#else
        __builtin_amdgcn_tensor_load_to_lds(g0, g1, g2, g3, 0);
#endif
        __builtin_amdgcn_s_wait_tensorcnt(0);
    }
#else
    {
        const uint4* src = (const uint4*)h1l;
        uint4* dst = (uint4*)sh_a;
        for (int e = tid; e < (B * H * 2) / 16; e += 256) dst[e] = src[e];
    }
#endif
    __syncthreads();

    // ---- GEMM2: h2 = lrelu(h1(256x128) @ W2(128x128) + b2), wave w owns M-tiles {2w,2w+1}
    for (int mi = 0; mi < 2; ++mi) {
        const int mt = wid * 2 + mi;
        const int m0 = mt * 16;
        v8f acc[8];
#pragma unroll
        for (int nt = 0; nt < 8; ++nt) acc[nt] = v8f_zero();
#pragma unroll
        for (int ks = 0; ks < 4; ++ks) {
            v16h a = frag_ld(sh_a, m0, ks * 32, lane);
#pragma unroll
            for (int nt = 0; nt < 8; ++nt) {
                v16h bf = frag_ld(sh_w2t, nt * 16, ks * 32, lane);
                acc[nt] = __builtin_amdgcn_wmma_f32_16x16x32_f16(
                    false, a, false, bf, (short)0, acc[nt], false, false);
            }
        }
        // bias + lrelu, write h2 back (f16) into the same rows of sh_a
        // (safe: this wave is the only reader/writer of rows [m0, m0+16))
#pragma unroll
        for (int nt = 0; nt < 8; ++nt) {
            const int n0 = nt * 16;
            const float bn = sh_b2[n0 + lr];
#pragma unroll
            for (int r = 0; r < 8; ++r) {
                int m = m0 + r + hb * 8;
                sh_a[m * H + n0 + lr] = (_Float16)lrelu(acc[nt][r] + bn);
            }
        }
    }
    __syncthreads();

    // ---- GEMM3: out = h2(256x128) @ Wout(128x32pad) + bout, + DCA epilogue ----
    const int plq = p * Q;
    for (int mi = 0; mi < 2; ++mi) {
        const int mt = wid * 2 + mi;
        const int m0 = mt * 16;
        v8f acc2[2];
        acc2[0] = v8f_zero(); acc2[1] = v8f_zero();
#pragma unroll
        for (int ks = 0; ks < 4; ++ks) {
            v16h a = frag_ld(sh_a, m0, ks * 32, lane);
#pragma unroll
            for (int nt = 0; nt < 2; ++nt) {
                v16h bf = frag_ld(sh_wo, nt * 16, ks * 32, lane);
                acc2[nt] = __builtin_amdgcn_wmma_f32_16x16x32_f16(
                    false, a, false, bf, (short)0, acc2[nt], false, false);
            }
        }
#pragma unroll
        for (int nt = 0; nt < 2; ++nt) {
            const int q = nt * 16 + lr;
            if (q < Q) {
                const float bq = sh_bo[q] + dca_bias[plq + q];
#pragma unroll
                for (int r = 0; r < 8; ++r) {
                    const int b = m0 + r + hb * 8;
                    float v = acc2[nt][r] + bq;
                    if (p < Q - 1) {  // masked DCA: col block p fed by row blocks p+1..20
                        const int* ib = idx + b * L;
                        for (int l2 = p + 1; l2 < Q; ++l2)
                            v += Wdca[(size_t)(l2 * Q + ib[l2]) * LQ + plq + q];
                    }
                    out[(size_t)b * L * Q + plq + q] = v;
                }
            }
        }
    }
}

// ---------------- kernel 3: output position 0 ----------------
__global__ void k_pos0(const float* __restrict__ w0, const float* __restrict__ b0,
                       const float* __restrict__ dca_bias, const float* __restrict__ Wdca,
                       const int* __restrict__ idx, float* __restrict__ out) {
    int t = blockIdx.x * blockDim.x + threadIdx.x;
    if (t >= B * Q) return;
    int b = t / Q, q = t % Q;
    float v = w0[q] + b0[q] + dca_bias[q];
    const int* ib = idx + b * L;
    for (int l2 = 1; l2 < Q; ++l2)
        v += Wdca[(size_t)(l2 * Q + ib[l2]) * LQ + q];
    out[(size_t)b * L * Q + q] = v;
}

extern "C" void kernel_launch(void* const* d_in, const int* in_sizes, int n_in,
                              void* d_out, int out_size, void* d_ws, size_t ws_size,
                              hipStream_t stream) {
    const float* x     = (const float*)d_in[0];
    const float* dbias = (const float*)d_in[1];
    const float* Wdca  = (const float*)d_in[2];
    const float* w0    = (const float*)d_in[3];
    const float* b0    = (const float*)d_in[4];
    const float* W1    = (const float*)d_in[5];
    const float* b1    = (const float*)d_in[6];
    const float* W2    = (const float*)d_in[7];
    const float* b2    = (const float*)d_in[8];
    const float* Wout  = (const float*)d_in[9];
    const float* bout  = (const float*)d_in[10];
    float* out = (float*)d_out;

    int* d_idx = (int*)d_ws;                                   // B*L ints = 200 KB
    _Float16* d_h1 = (_Float16*)((char*)d_ws + (size_t)B * L * sizeof(int)); // 13 MB

    k_idx<<<(B * L + 255) / 256, 256, 0, stream>>>(x, d_idx);
    k_h1<<<dim3(B, L - 1), H, 0, stream>>>(W1, b1, d_idx, d_h1);

    const size_t smem = 65536 + 32768 + 8192 + 512 + 128;       // ~107 KB of 320 KB/WGP
    k_ffn<<<L - 1, 256, smem, stream>>>(d_h1, W2, b2, Wout, bout, Wdca, dbias, d_idx, out);

    k_pos0<<<(B * Q + 255) / 256, 256, 0, stream>>>(w0, b0, dbias, Wdca, d_idx, out);
}
